// StackedGatPerfPlayerModel_18141941858959
// MI455X (gfx1250) — matrix operand order repaired and checked by reading, not compiled, not run on hardware
//
#include <hip/hip_runtime.h>
#include <hip/hip_bf16.h>

#define BB 8
#define NN 4096
#define DEG 16
#define DIN 64
#define DM 128
#define HH 4
#define DH 32
#define DO 32
#define NLAYERS 3
#define NNODES (BB*NN)          // 32768 rows
#define NN_SCALE 1999853.335557038f

typedef __attribute__((ext_vector_type(16))) __bf16 v16bf;
typedef __attribute__((ext_vector_type(8)))  float  v8f;

__device__ __forceinline__ float elu_f(float a) {
    return a > 0.f ? a : (__expf(a) - 1.f);
}

__device__ __forceinline__ unsigned int pack_bf16x2(float f0, float f1) {
    __bf16 h0 = (__bf16)f0, h1 = (__bf16)f1;
    unsigned short u0 = __builtin_bit_cast(unsigned short, h0);
    unsigned short u1 = __builtin_bit_cast(unsigned short, h1);
    return (unsigned int)u0 | ((unsigned int)u1 << 16);
}

// ---------------------------------------------------------------------------
// WMMA GEMM: C[M x 128] = act(A[M x K] @ Bm[K x 128] + bias)
// Block = 256 threads = 8 waves; block tile = 16 rows x 128 cols, wave w owns
// columns [16w,16w+16). K stepped by 32 via v_wmma_f32_16x16x32_bf16.
// Tiles are staged in LDS in *fragment order* so each lane reads its whole
// 16-element bf16 fragment with one 32-byte vector load (2x ds_load_b128).
//
// A-frag layout (16x32 bf16, wave32): lane = m + 16*hh, element pair v holds
//   k = (v<4 ? 2v : 16+2(v-4)) + 8*hh  and k+1.
// B-frag layout (32x16 bf16): lane = n_local + 16*hh, pair v holds
//   k = (v<4 ? 2v : 16+2(v-4)) + 8*hh and k+1 (same order as A).
// ---------------------------------------------------------------------------
template<int K, bool HAS_BIAS, bool ACT>
__global__ __launch_bounds__(256)
void gemm128_kernel(const float* __restrict__ A, const float* __restrict__ Bm,
                    const float* __restrict__ bias, float* __restrict__ C) {
    __shared__ __align__(32) unsigned int asw32[32 * 8];       // [lane][pair]
    __shared__ __align__(32) unsigned int bsw32[8 * 32 * 8];   // [wave][lane][pair]
    const int t    = threadIdx.x;
    const int lane = t & 31;
    const int w    = t >> 5;
    const int rowBase = blockIdx.x * 16;

    v8f acc = {};

    #pragma unroll
    for (int ks = 0; ks < K; ks += 32) {
        // ---- stage A tile (16x32) in fragment order: 256 k-pairs, 1/thread
        {
            int m  = t >> 4;            // 0..15
            int ke = (t & 15) * 2;      // even k: 0..30
            const float* ap = A + (size_t)(rowBase + m) * K + ks + ke;
            float f0 = ap[0], f1 = ap[1];
            int hh2 = (ke >> 3) & 1;            // 8..15 / 24..31 -> upper half-lane
            int kk  = ke - 8 * hh2;             // {0,2,4,6} u {16,18,20,22}
            int v   = (kk < 8) ? (kk >> 1) : (4 + ((kk - 16) >> 1));
            asw32[(m + 16 * hh2) * 8 + v] = pack_bf16x2(f0, f1);
        }
        // ---- stage B tile (32x128) in fragment order: 2048 k-pairs, 8/thread
        #pragma unroll
        for (int i = 0; i < 8; i++) {
            int p  = t + 256 * i;       // 0..2047
            int ke = (p >> 7) * 2;      // even k: 0..30
            int c  = p & 127;           // column
            float f0 = Bm[(size_t)(ks + ke)     * 128 + c];
            float f1 = Bm[(size_t)(ks + ke + 1) * 128 + c];
            int wv  = c >> 4;
            int nl  = c & 15;
            int hh2 = (ke >> 3) & 1;    // 0 or 1
            int v   = ((ke - 8 * hh2) < 8) ? ((ke - 8 * hh2) >> 1) : (4 + ((ke - 8 * hh2 - 16) >> 1));
            bsw32[(wv * 32 + nl + 16 * hh2) * 8 + v] = pack_bf16x2(f0, f1);
        }
        __syncthreads();

        v16bf afrag = ((const v16bf*)asw32)[lane];
        v16bf bfrag = ((const v16bf*)bsw32)[w * 32 + lane];
        acc = __builtin_amdgcn_wmma_f32_16x16x32_bf16(
                  false, afrag, false, bfrag, (short)0, acc, false, false);
        __syncthreads();
    }

    // ---- epilogue: C/D layout -> VGPR r: lanes 0-15 M=r, lanes 16-31 M=r+8
    const int ncol = 16 * w + (lane & 15);
    const int mofs = 8 * (lane >> 4);
    float bv = HAS_BIAS ? bias[ncol] : 0.f;
    #pragma unroll
    for (int r = 0; r < 8; r++) {
        float val = acc[r] + bv;
        if (ACT) val = elu_f(val);
        C[(size_t)(rowBase + r + mofs) * 128 + ncol] = val;
    }
}

// ---------------------------------------------------------------------------
// Repack layer weights: Wcol[d][c] = Wl[l][c/32][d][c%32]  (128x128)
//                       Wblk[kk][c] = (same head) ? Wo[l][h][k][o] : 0
// ---------------------------------------------------------------------------
__global__ __launch_bounds__(256)
void prep_weights_kernel(const float* __restrict__ Wl, const float* __restrict__ Wo,
                         int l, float* __restrict__ Wcol, float* __restrict__ Wblk) {
    int t = blockIdx.x * 256 + threadIdx.x;   // 0..32767
    if (t < 16384) {
        int d = t >> 7, c = t & 127;
        Wcol[t] = Wl[(size_t)l * HH * DM * DH + (size_t)(c >> 5) * DM * DH + d * DH + (c & 31)];
    } else {
        int i  = t - 16384;
        int kk = i >> 7, c = i & 127;
        float v = 0.f;
        if ((kk >> 5) == (c >> 5))
            v = Wo[(size_t)l * HH * DH * DO + (size_t)(kk >> 5) * DH * DO + (kk & 31) * DO + (c & 31)];
        Wblk[i] = v;
    }
}

// ---------------------------------------------------------------------------
// Attention: one thread per (b, n, head). Gathers DEG=16 neighbor rows of h
// (L2-resident), leaky-relu + mask + softmax over neighbors, aggregates.
// ---------------------------------------------------------------------------
__global__ __launch_bounds__(256)
void attn_kernel(const float* __restrict__ h, const int* __restrict__ adj,
                 const float* __restrict__ masks,
                 const float* __restrict__ a_src, const float* __restrict__ a_dst,
                 float* __restrict__ agg) {
    __shared__ float s_asrc[HH * DH], s_adst[HH * DH];
    int t = threadIdx.x;
    if (t < HH * DH) { s_asrc[t] = a_src[t]; s_adst[t] = a_dst[t]; }
    __syncthreads();

    int gid = blockIdx.x * 256 + t;        // (b*N + n)*H + head
    int hd  = gid & (HH - 1);
    int bn  = gid >> 2;                    // b*N + n
    int b   = bn >> 12;                    // N = 4096

    const float* hrow = h + (size_t)bn * DM + hd * DH;
    float esrc = 0.f;
    #pragma unroll
    for (int k = 0; k < DH; k++) esrc += hrow[k] * s_asrc[hd * DH + k];

    float e[DEG];
    float emax = -1e30f;
    #pragma unroll 4
    for (int m = 0; m < DEG; m++) {
        int nb = adj[(size_t)bn * DEG + m];
        const float* hn = h + ((size_t)(b * NN + nb)) * DM + hd * DH;
        float ed = 0.f;
        #pragma unroll
        for (int k = 0; k < DH; k++) ed += hn[k] * s_adst[hd * DH + k];
        float ee = esrc + ed;
        ee = ee > 0.f ? ee : 0.2f * ee;                 // leaky_relu 0.2
        if (masks[(size_t)bn * DEG + m] <= 0.f) ee = -1e9f;
        e[m] = ee;
        emax = fmaxf(emax, ee);
    }
    float ssum = 0.f;
    #pragma unroll
    for (int m = 0; m < DEG; m++) { e[m] = __expf(e[m] - emax); ssum += e[m]; }
    float inv = 1.f / ssum;

    float accv[DH];
    #pragma unroll
    for (int k = 0; k < DH; k++) accv[k] = 0.f;
    #pragma unroll 4
    for (int m = 0; m < DEG; m++) {
        int nb = adj[(size_t)bn * DEG + m];
        const float* hn = h + ((size_t)(b * NN + nb)) * DM + hd * DH;
        float al = e[m] * inv;
        #pragma unroll
        for (int k = 0; k < DH; k++) accv[k] += al * hn[k];
    }
    float* outp = agg + (size_t)bn * DM + hd * DH;
    #pragma unroll
    for (int k = 0; k < DH; k++) outp[k] = accv[k];
}

// ---------------------------------------------------------------------------
// x = LayerNorm(out + x) * g + b   — one wave32 per 128-wide row
// ---------------------------------------------------------------------------
__global__ __launch_bounds__(128)
void residual_ln_kernel(const float* __restrict__ out, float* __restrict__ x,
                        const float* __restrict__ g, const float* __restrict__ bta) {
    int row  = blockIdx.x * 4 + (threadIdx.x >> 5);
    int lane = threadIdx.x & 31;
    const float* o  = out + (size_t)row * DM;
    float*       xr = x   + (size_t)row * DM;
    float v[4], s = 0.f, s2 = 0.f;
    #pragma unroll
    for (int i = 0; i < 4; i++) {
        int c = lane + 32 * i;
        v[i] = o[c] + xr[c];
        s += v[i]; s2 += v[i] * v[i];
    }
    #pragma unroll
    for (int off = 16; off > 0; off >>= 1) {
        s  += __shfl_xor(s,  off, 32);
        s2 += __shfl_xor(s2, off, 32);
    }
    float mu  = s * (1.f / 128.f);
    float var = s2 * (1.f / 128.f) - mu * mu;
    float rs  = rsqrtf(var + 1e-5f);
    #pragma unroll
    for (int i = 0; i < 4; i++) {
        int c = lane + 32 * i;
        xr[c] = (v[i] - mu) * rs * g[c] + bta[c];
    }
}

// ---------------------------------------------------------------------------
// Final 128 -> 256 -> 128 -> 8 MLP on the gathered query rows (one block / b)
// ---------------------------------------------------------------------------
__global__ __launch_bounds__(256)
void final_mlp_kernel(const float* __restrict__ x, const int* __restrict__ qidx,
                      const float* __restrict__ Wf1, const float* __restrict__ bf1,
                      const float* __restrict__ Wf2, const float* __restrict__ bf2,
                      const float* __restrict__ Wf3, const float* __restrict__ bf3,
                      float* __restrict__ outp) {
    __shared__ float v0[128], v1[256], v2[128];
    int b = blockIdx.x, t = threadIdx.x;
    int q = qidx[b];
    if (t < 128) v0[t] = x[((size_t)b * NN + q) * DM + t];
    __syncthreads();
    {
        float a = bf1[t];
        for (int d = 0; d < 128; d++) a += v0[d] * Wf1[d * 256 + t];
        v1[t] = elu_f(a);
    }
    __syncthreads();
    if (t < 128) {
        float a = bf2[t];
        for (int d = 0; d < 256; d++) a += v1[d] * Wf2[d * 128 + t];
        v2[t] = elu_f(a);
    }
    __syncthreads();
    if (t < 8) {
        float a = bf3[t];
        for (int d = 0; d < 128; d++) a += v2[d] * Wf3[d * 8 + t];
        outp[b * 8 + t] = elu_f(a) * NN_SCALE;
    }
}

// ---------------------------------------------------------------------------
extern "C" void kernel_launch(void* const* d_in, const int* in_sizes, int n_in,
                              void* d_out, int out_size, void* d_ws, size_t ws_size,
                              hipStream_t stream) {
    const float* nf    = (const float*)d_in[0];
    const int*   qidx  = (const int*)  d_in[1];
    const float* masks = (const float*)d_in[2];
    const int*   adj   = (const int*)  d_in[3];
    // d_in[4] = sim_results (unused by reference)
    const float* W0    = (const float*)d_in[5];
    const float* b0    = (const float*)d_in[6];
    const float* Wl    = (const float*)d_in[7];
    const float* a_src = (const float*)d_in[8];
    const float* a_dst = (const float*)d_in[9];
    const float* Wo    = (const float*)d_in[10];
    const float* Wlin  = (const float*)d_in[11];
    const float* blin  = (const float*)d_in[12];
    const float* ln_g  = (const float*)d_in[13];
    const float* ln_b  = (const float*)d_in[14];
    const float* Wf1   = (const float*)d_in[15];
    const float* bf1   = (const float*)d_in[16];
    const float* Wf2   = (const float*)d_in[17];
    const float* bf2   = (const float*)d_in[18];
    const float* Wf3   = (const float*)d_in[19];
    const float* bf3   = (const float*)d_in[20];
    float* outp = (float*)d_out;

    const size_t ROWF = (size_t)NNODES * DM;   // 4,194,304 floats
    float* ws   = (float*)d_ws;
    float* x    = ws;                 // 16 MB
    float* h    = ws + ROWF;          // 16 MB
    float* agg  = ws + 2 * ROWF;      // 16 MB (reused in-place for Wo/Wlin GEMMs)
    float* Wcol = ws + 3 * ROWF;      // 64 KB
    float* Wblk = Wcol + DM * DM;     // 64 KB

    const int GEMM_GRID = NNODES / 16;   // 2048

    // x = elu(nf @ W0 + b0)
    gemm128_kernel<DIN, true, true><<<GEMM_GRID, 256, 0, stream>>>(nf, W0, b0, x);

    for (int l = 0; l < NLAYERS; l++) {
        prep_weights_kernel<<<128, 256, 0, stream>>>(Wl, Wo, l, Wcol, Wblk);
        // h = x @ Wcol
        gemm128_kernel<DM, false, false><<<GEMM_GRID, 256, 0, stream>>>(x, Wcol, nullptr, h);
        // attention + aggregation
        attn_kernel<<<(NNODES * HH) / 256, 256, 0, stream>>>(
            h, adj, masks, a_src + (size_t)l * HH * DH, a_dst + (size_t)l * HH * DH, agg);
        // per-head output projection as block-diag GEMM (in place)
        gemm128_kernel<DM, false, false><<<GEMM_GRID, 256, 0, stream>>>(agg, Wblk, nullptr, agg);
        // out = elu(agg @ Wlin + blin) (in place)
        gemm128_kernel<DM, true, true><<<GEMM_GRID, 256, 0, stream>>>(
            agg, Wlin + (size_t)l * DM * DM, blin + (size_t)l * DM, agg);
        // x = LN(out + x)
        residual_ln_kernel<<<NNODES / 4, 128, 0, stream>>>(
            agg, x, ln_g + (size_t)l * DM, ln_b + (size_t)l * DM);
    }

    final_mlp_kernel<<<BB, 256, 0, stream>>>(x, qidx, Wf1, bf1, Wf2, bf2, Wf3, bf3, outp);
}
